// GNN_75282186764453
// MI455X (gfx1250) — matrix operand order repaired and checked by reading, not compiled, not run on hardware
//
#include <hip/hip_runtime.h>
#include <hip/hip_bf16.h>

// ---------------- problem constants (from reference) ----------------
#define HH    4          // heads
#define CC    64         // per-head dim
#define HCHC  256        // H*C
#define BG    512        // graphs
#define NPG0  128        // nodes per graph (initial)
#define EE    262144     // edges
#define N0    65536      // B*NPG
#define FIN   30
#define EDIM  11
#define LNUM  3

typedef __attribute__((ext_vector_type(16))) _Float16 v16h;
typedef __attribute__((ext_vector_type(8)))  float    v8f;

// ---------------- helpers ----------------
__device__ __forceinline__ unsigned ordEnc(float f) {
    unsigned u = __float_as_uint(f);
    return (u >> 31) ? ~u : (u | 0x80000000u);
}
__device__ __forceinline__ float ordDec(unsigned u) {
    return (u >> 31) ? __uint_as_float(u ^ 0x80000000u) : __uint_as_float(~u);
}

// async global->LDS 16B copy (CDNA5); completion via s_wait_asynccnt + barrier
__device__ __forceinline__ void asyncCopy16(void* ldsPtr, const void* gPtr) {
    unsigned lds = (unsigned)(unsigned long long)ldsPtr;
    asm volatile("global_load_async_to_lds_b128 %0, %1, off"
                 :: "v"(lds), "v"((unsigned long long)gPtr) : "memory");
}
__device__ __forceinline__ void asyncWait() {
    asm volatile("s_wait_asynccnt 0" ::: "memory");
}

__global__ void zerofill_kernel(unsigned* p, long n) {
    long i = (long)blockIdx.x * blockDim.x + threadIdx.x;
    if (i < n) p[i] = 0u;
}

// ---------------- weight pre-swizzle: f32 [K,Nc] -> f16 fragment order -------
// Layout: [n>>6][k>>4][n&63][k&15]  (per 64-col block: contiguous panels,
// each lane's B fragment = 32 contiguous bytes)
__global__ void wswizzle_kernel(const float* __restrict__ W, _Float16* __restrict__ Wf16,
                                int K, int Nc)
{
    int idx = blockIdx.x * 256 + threadIdx.x;
    if (idx >= K * Nc) return;
    int k = idx / Nc, n = idx % Nc;
    int npan = (K + 15) >> 4;
    long d = (((long)(n >> 6) * npan + (k >> 4)) * 64 + (n & 63)) * 16 + (k & 15);
    Wf16[d] = (_Float16)W[idx];
}

// ---------------- WMMA GEMM: Out[M,N] = A[M,K] @ W[K,N] + bias (opt relu) ----
// Block = 256 thr (8 waves), block tile = 128(M) x 64(N).
// Pre-swizzled f16 weight tile DMA'd into LDS via global_load_async_to_lds_b128.
// Each wave: 16x64 strip = 4 accumulators; 4 WMMA per 32-K step.
__global__ void wmma_gemm_kernel(const float* __restrict__ A,
                                 const _Float16* __restrict__ Wf16,
                                 const float* __restrict__ bias,
                                 float* __restrict__ Out,
                                 int M, int K, int Nc, int doRelu)
{
    __shared__ alignas(32) _Float16 Wlds[256 * 64];  // up to K=256: 32KB
    int tid  = threadIdx.x;
    int lane = tid & 31;
    int wave = tid >> 5;
    int mBlocks = M >> 7;
    int bm = blockIdx.x % mBlocks;
    int bn = blockIdx.x / mBlocks;
    int n0 = bn << 6;

    int npan = (K + 15) >> 4;
    const char* wsrc = (const char*)(Wf16 + (size_t)bn * npan * 1024);
    int bytes = npan * 2048;
    for (int o = tid * 16; o < bytes; o += 4096)
        asyncCopy16((char*)Wlds + o, wsrc + o);
    asyncWait();
    __syncthreads();

    int tm   = (bm << 7) + (wave << 4);
    int rowA = tm + (lane & 15);
    int kbA  = (lane < 16) ? 0 : 16;     // A: half-wave K block
    int khB  = (lane < 16) ? 0 : 16;     // B: half-wave K block
    int colL = lane & 15;
    const float* Arow = A + (size_t)rowA * K;

    v8f c[4] = {v8f{}, v8f{}, v8f{}, v8f{}};
    for (int k0 = 0; k0 < K; k0 += 32) {
        v16h a;
        if (k0 + 32 <= K) {
            if (k0 + 64 <= K) __builtin_prefetch(Arow + k0 + 32, 0, 3);
            const float2* A2 = (const float2*)(Arow + k0);
#pragma unroll
            for (int j = 0; j < 4; ++j) {
                float2 p0 = A2[(kbA + 2 * j) >> 1];
                float2 p1 = A2[(8 + kbA + 2 * j) >> 1];
                a[2 * j]     = (_Float16)p0.x;  a[2 * j + 1] = (_Float16)p0.y;
                a[8 + 2 * j] = (_Float16)p1.x;  a[9 + 2 * j] = (_Float16)p1.y;
            }
        } else {                          // ragged tail (K=30)
#pragma unroll
            for (int j = 0; j < 4; ++j) {
                int k1 = k0 + kbA + 2 * j, k2 = k0 + 8 + kbA + 2 * j;
                a[2 * j]     = (k1 < K)     ? (_Float16)Arow[k1]     : (_Float16)0.f;
                a[2 * j + 1] = (k1 + 1 < K) ? (_Float16)Arow[k1 + 1] : (_Float16)0.f;
                a[8 + 2 * j] = (k2 < K)     ? (_Float16)Arow[k2]     : (_Float16)0.f;
                a[9 + 2 * j] = (k2 + 1 < K) ? (_Float16)Arow[k2 + 1] : (_Float16)0.f;
            }
        }
        int panel = (k0 + khB) >> 4;
        v16h bf[4];
#pragma unroll
        for (int nt = 0; nt < 4; ++nt)
            bf[nt] = *(const v16h*)&Wlds[((panel * 64) + (nt << 4) + colL) << 4];
#pragma unroll
        for (int nt = 0; nt < 4; ++nt)
            c[nt] = __builtin_amdgcn_wmma_f32_16x16x32_f16(
                        false, a, false, bf[nt], (short)0, c[nt], false, false);
    }

    int mbase = tm + ((lane < 16) ? 0 : 8);
#pragma unroll
    for (int nt = 0; nt < 4; ++nt) {
        int col = n0 + (nt << 4) + colL;
        float bv = bias ? bias[col] : 0.f;
#pragma unroll
        for (int i = 0; i < 8; ++i) {
            float v = c[nt][i] + bv;
            if (doRelu) v = fmaxf(v, 0.f);
            Out[(size_t)(mbase + i) * Nc + col] = v;
        }
    }
}

// ---------------- attention score: a[e,h] = q[dst]·(k[src]+ee)/8 -------------
__global__ void attn_score_kernel(const float* __restrict__ qb, const float* __restrict__ kb,
                                  const float* __restrict__ ea, const float* __restrict__ We,
                                  const int* __restrict__ src, const int* __restrict__ dst,
                                  const int* __restrict__ mask,
                                  float* __restrict__ aE, unsigned* __restrict__ amax, int E_)
{
    __shared__ alignas(16) float sWe[EDIM * HCHC];   // 11 KB, async DMA'd
    for (int o = threadIdx.x * 16; o < EDIM * HCHC * 4; o += 4096)
        asyncCopy16((char*)sWe + o, (const char*)We + o);
    asyncWait();
    __syncthreads();
    int idx = blockIdx.x * 256 + threadIdx.x;
    if (idx >= E_ * HH) return;
    int e = idx >> 2, h = idx & 3;
    if (mask && !mask[e]) { aE[idx] = -1e30f; return; }
    int s = src[e], d = dst[e];
    const float4* qp = (const float4*)(qb + (size_t)d * HCHC + h * CC);
    const float4* kp = (const float4*)(kb + (size_t)s * HCHC + h * CC);
    const float* eap = ea + (size_t)e * EDIM;
    float ec[EDIM];
#pragma unroll
    for (int t = 0; t < EDIM; ++t) ec[t] = eap[t];
    float acc = 0.f;
#pragma unroll 4
    for (int c4 = 0; c4 < 16; ++c4) {
        float4 q4 = qp[c4];
        float4 k4 = kp[c4];
        float ex = 0.f, ey = 0.f, ez = 0.f, ew = 0.f;
#pragma unroll
        for (int t = 0; t < EDIM; ++t) {
            float4 w4 = *(const float4*)&sWe[t * HCHC + h * CC + (c4 << 2)];
            ex += ec[t] * w4.x; ey += ec[t] * w4.y;
            ez += ec[t] * w4.z; ew += ec[t] * w4.w;
        }
        acc += q4.x * (k4.x + ex) + q4.y * (k4.y + ey)
             + q4.z * (k4.z + ez) + q4.w * (k4.w + ew);
    }
    acc *= 0.125f;            // 1/sqrt(64)
    aE[idx] = acc;
    atomicMax(&amax[(size_t)d * HH + h], ordEnc(acc));
}

// exp(a - max) in place, accumulate denominator
__global__ void attn_exp_kernel(float* __restrict__ wE, const unsigned* __restrict__ amax,
                                const int* __restrict__ dst, const int* __restrict__ mask,
                                float* __restrict__ den, int E_)
{
    int idx = blockIdx.x * blockDim.x + threadIdx.x;
    if (idx >= E_ * HH) return;
    int e = idx >> 2, h = idx & 3;
    if (mask && !mask[e]) { wE[idx] = 0.f; return; }
    float mx = ordDec(amax[(size_t)dst[e] * HH + h]);
    float ex = __expf(wE[idx] - mx);
    wE[idx] = ex;
    atomicAdd(&den[(size_t)dst[e] * HH + h], ex);
}

__global__ void attn_norm_kernel(float* __restrict__ wE, const float* __restrict__ den,
                                 const int* __restrict__ dst, int E_)
{
    int idx = blockIdx.x * blockDim.x + threadIdx.x;
    if (idx >= E_ * HH) return;
    int e = idx >> 2, h = idx & 3;
    wE[idx] = wE[idx] / fmaxf(den[(size_t)dst[e] * HH + h], 1e-16f);
}

__global__ void attn_mean_kernel(const float* __restrict__ wE, float* __restrict__ am, int E_)
{
    int e = blockIdx.x * blockDim.x + threadIdx.x;
    if (e >= E_) return;
    am[e] = 0.25f * (wE[e * 4] + wE[e * 4 + 1] + wE[e * 4 + 2] + wE[e * 4 + 3]);
}

// agg[dst,hc] += (v[src,hc] + ee(e,hc)) * w[e,h]; thread = (edge, 4 channels)
__global__ void aggregate_kernel(const float* __restrict__ vb, const float* __restrict__ ea,
                                 const float* __restrict__ We, const float* __restrict__ wE,
                                 const int* __restrict__ src, const int* __restrict__ dst,
                                 const int* __restrict__ mask, float* __restrict__ agg, int E_)
{
    int idx = blockIdx.x * 256 + threadIdx.x;
    if (idx >= E_ * 64) return;
    int e = idx >> 6, c4 = (idx & 63) << 2;
    if (mask && !mask[e]) return;
    float w = wE[(size_t)e * HH + (c4 >> 6)];
    const float* eap = ea + (size_t)e * EDIM;
    float ex = 0.f, ey = 0.f, ez = 0.f, ew = 0.f;
#pragma unroll
    for (int t = 0; t < EDIM; ++t) {
        float et = eap[t];
        float4 w4 = *(const float4*)&We[t * HCHC + c4];
        ex += et * w4.x; ey += et * w4.y; ez += et * w4.z; ew += et * w4.w;
    }
    float4 v4 = *(const float4*)&vb[(size_t)src[e] * HCHC + c4];
    float* dp = &agg[(size_t)dst[e] * HCHC + c4];
    atomicAdd(dp + 0, (v4.x + ex) * w);
    atomicAdd(dp + 1, (v4.y + ey) * w);
    atomicAdd(dp + 2, (v4.z + ez) * w);
    atomicAdd(dp + 3, (v4.w + ew) * w);
}

// beta gate + mix (writes hmix in place over agg)
__global__ void betamix_kernel(float* __restrict__ agg, const float* __restrict__ skip,
                               const float* __restrict__ Wb, int n)
{
    int i = blockIdx.x * blockDim.x + threadIdx.x;
    if (i >= n) return;
    float* ag = agg + (size_t)i * HCHC;
    const float* sk = skip + (size_t)i * HCHC;
    float acc = 0.f;
    for (int c4 = 0; c4 < HCHC; c4 += 4) {
        float4 a4 = *(const float4*)&ag[c4];
        float4 s4 = *(const float4*)&sk[c4];
        float4 w1 = *(const float4*)&Wb[c4];
        float4 w2 = *(const float4*)&Wb[HCHC + c4];
        float4 w3 = *(const float4*)&Wb[2 * HCHC + c4];
        acc += a4.x * w1.x + s4.x * w2.x + (a4.x - s4.x) * w3.x;
        acc += a4.y * w1.y + s4.y * w2.y + (a4.y - s4.y) * w3.y;
        acc += a4.z * w1.z + s4.z * w2.z + (a4.z - s4.z) * w3.z;
        acc += a4.w * w1.w + s4.w * w2.w + (a4.w - s4.w) * w3.w;
    }
    float beta = 1.f / (1.f + __expf(-acc));
    for (int c4 = 0; c4 < HCHC; c4 += 4) {
        float4 a4 = *(const float4*)&ag[c4];
        float4 s4 = *(const float4*)&sk[c4];
        float4 r;
        r.x = beta * s4.x + (1.f - beta) * a4.x;
        r.y = beta * s4.y + (1.f - beta) * a4.y;
        r.z = beta * s4.z + (1.f - beta) * a4.z;
        r.w = beta * s4.w + (1.f - beta) * a4.w;
        *(float4*)&ag[c4] = r;
    }
}

// BN stats (sum, sumsq) per channel of h[n,64]
__global__ void bn_stats_kernel(const float* __restrict__ h, float* __restrict__ stats, int n)
{
    int tid = blockIdx.x * blockDim.x + threadIdx.x;
    int c = tid & 63;
    int r0 = tid >> 6;
    int stride = (gridDim.x * blockDim.x) >> 6;
    float s = 0.f, sq = 0.f;
    for (int r = r0; r < n; r += stride) {
        float v = h[(size_t)r * CC + c];
        s += v; sq += v * v;
    }
    atomicAdd(&stats[c], s);
    atomicAdd(&stats[CC + c], sq);
}

__global__ void bn_apply_kernel(float* __restrict__ h, const float* __restrict__ stats,
                                const float* __restrict__ g, const float* __restrict__ b, int n)
{
    int idx = blockIdx.x * blockDim.x + threadIdx.x;
    if (idx >= n * CC) return;
    int c = idx & 63;
    float inv = 1.f / (float)n;
    float m = stats[c] * inv;
    float v = stats[CC + c] * inv - m * m;
    h[idx] = (h[idx] - m) * rsqrtf(v + 1e-5f) * g[c] + b[c];
}

// ---------------- TopK pooling ----------------
__global__ void pnorm_kernel(const float* __restrict__ P, float* __restrict__ pn)
{
    __shared__ float sh[CC];
    sh[threadIdx.x] = P[threadIdx.x] * P[threadIdx.x];
    __syncthreads();
    if (threadIdx.x == 0) {
        float s = 0.f;
        for (int i = 0; i < CC; ++i) s += sh[i];
        *pn = sqrtf(s);
    }
}

__global__ void score_kernel(const float* __restrict__ h, const float* __restrict__ P,
                             const float* __restrict__ pn, float* __restrict__ score,
                             int* __restrict__ remap, int n)
{
    int i = blockIdx.x * blockDim.x + threadIdx.x;
    if (i >= n) return;
    float inv = 1.f / (*pn);
    float s = 0.f;
    for (int c = 0; c < CC; ++c) s += h[(size_t)i * CC + c] * P[c];
    score[i] = s * inv;
    remap[i] = -1;
}

// one block per graph; rank by score (desc, stable), keep rank<k
__global__ void topk_kernel(const float* __restrict__ h, const float* __restrict__ score,
                            float* __restrict__ hnew, int* __restrict__ remap, int npg, int k)
{
    __shared__ float sc[NPG0];
    int g = blockIdx.x, i = threadIdx.x;
    if (i < npg) sc[i] = score[g * npg + i];
    __syncthreads();
    if (i >= npg) return;
    float si = sc[i];
    int rank = 0;
    for (int j = 0; j < npg; ++j) {
        float sj = sc[j];
        rank += (sj > si) || (sj == si && j < i);
    }
    if (rank < k) {
        int old = g * npg + i;
        int nid = g * k + rank;
        float t = tanhf(si);
        for (int c = 0; c < CC; ++c)
            hnew[(size_t)nid * CC + c] = h[(size_t)old * CC + c] * t;
        remap[old] = nid;
    }
}

__global__ void remap_edges_kernel(const int* __restrict__ src, const int* __restrict__ dst,
                                   const int* __restrict__ mask, const int* __restrict__ remap,
                                   int* __restrict__ nsrc, int* __restrict__ ndst,
                                   int* __restrict__ nmask, int E_)
{
    int e = blockIdx.x * blockDim.x + threadIdx.x;
    if (e >= E_) return;
    int ms = mask ? mask[e] : 1;
    int a = remap[src[e]], b = remap[dst[e]];
    int valid = (a >= 0) && (b >= 0) && ms;
    nsrc[e] = valid ? a : 0;
    ndst[e] = valid ? b : 0;
    nmask[e] = valid;
}

// per-graph max/mean readout, accumulated into grep[B,128]
__global__ void readout_kernel(const float* __restrict__ h, float* __restrict__ grep, int npg)
{
    int idx = blockIdx.x * blockDim.x + threadIdx.x;
    if (idx >= BG * CC) return;
    int g = idx >> 6, c = idx & 63;
    float mx = -1e30f, sm = 0.f;
    for (int r = 0; r < npg; ++r) {
        float v = h[(size_t)(g * npg + r) * CC + c];
        mx = fmaxf(mx, v);
        sm += v;
    }
    grep[g * 2 * CC + c]      += mx;
    grep[g * 2 * CC + CC + c] += sm / (float)npg;
}

__global__ void finalize_kernel(const float* __restrict__ grep, const int* __restrict__ eidx,
                                const float* __restrict__ attnm, float* __restrict__ out)
{
    int i = blockIdx.x * blockDim.x + threadIdx.x;
    const int n1 = BG * 2 * CC;            // 65536
    const int n2 = n1 + 2 * EE;            // + 524288
    const int n3 = n2 + EE;                // + 262144
    if (i >= n3) return;
    if (i < n1)       out[i] = fmaxf(grep[i], 0.f);
    else if (i < n2)  ((int*)out)[i] = eidx[i - n1];
    else              out[i] = attnm[i - n2];
}

// ---------------- host orchestration ----------------
static inline char* wsAlloc(void* ws, size_t& off, size_t bytes) {
    char* p = (char*)ws + off;
    off += (bytes + 255) & ~(size_t)255;
    return p;
}
#define ZERO(ptr, nwords) zerofill_kernel<<<((long)(nwords) + 255) / 256, 256, 0, stream>>>((unsigned*)(ptr), (long)(nwords))

extern "C" void kernel_launch(void* const* d_in, const int* in_sizes, int n_in,
                              void* d_out, int out_size, void* d_ws, size_t ws_size,
                              hipStream_t stream)
{
    const float* x    = (const float*)d_in[0];
    const float* ea   = (const float*)d_in[1];
    const int*   eidx = (const int*)d_in[2];
    const float* Wq0 = (const float*)d_in[4];  const float* bq0 = (const float*)d_in[5];
    const float* Wk0 = (const float*)d_in[6];  const float* bk0 = (const float*)d_in[7];
    const float* Wv0 = (const float*)d_in[8];  const float* bv0 = (const float*)d_in[9];
    const float* We0 = (const float*)d_in[10];
    const float* Ws0 = (const float*)d_in[11]; const float* bs0 = (const float*)d_in[12];
    const float* Wb0 = (const float*)d_in[13];
    const float* Wt0 = (const float*)d_in[14]; const float* bt0 = (const float*)d_in[15];
    const float* g0  = (const float*)d_in[16]; const float* b0  = (const float*)d_in[17];
    const float* Wq  = (const float*)d_in[18]; const float* bq  = (const float*)d_in[19];
    const float* Wk  = (const float*)d_in[20]; const float* bk  = (const float*)d_in[21];
    const float* Wv  = (const float*)d_in[22]; const float* bv  = (const float*)d_in[23];
    const float* We  = (const float*)d_in[24];
    const float* Ws  = (const float*)d_in[25]; const float* bs  = (const float*)d_in[26];
    const float* Wb  = (const float*)d_in[27];
    const float* Wt  = (const float*)d_in[28]; const float* bt  = (const float*)d_in[29];
    const float* gg  = (const float*)d_in[30]; const float* bb  = (const float*)d_in[31];
    const float* P   = (const float*)d_in[32];

    size_t off = 0;
    float* qb   = (float*)wsAlloc(d_ws, off, (size_t)N0 * HCHC * 4);
    float* kb   = (float*)wsAlloc(d_ws, off, (size_t)N0 * HCHC * 4);
    float* vb   = (float*)wsAlloc(d_ws, off, (size_t)N0 * HCHC * 4);
    float* sb   = (float*)wsAlloc(d_ws, off, (size_t)N0 * HCHC * 4);
    float* aggb = (float*)wsAlloc(d_ws, off, (size_t)N0 * HCHC * 4);
    float* h64a = (float*)wsAlloc(d_ws, off, (size_t)N0 * CC * 4);
    float* h64b = (float*)wsAlloc(d_ws, off, (size_t)N0 * CC * 4);
    float* wE   = (float*)wsAlloc(d_ws, off, (size_t)EE * HH * 4);
    unsigned* amax = (unsigned*)wsAlloc(d_ws, off, (size_t)N0 * HH * 4);
    float* den  = (float*)wsAlloc(d_ws, off, (size_t)N0 * HH * 4);
    float* score = (float*)wsAlloc(d_ws, off, (size_t)N0 * 4);
    int*   remap = (int*)wsAlloc(d_ws, off, (size_t)N0 * 4);
    int* srcA  = (int*)wsAlloc(d_ws, off, (size_t)EE * 4);
    int* dstA  = (int*)wsAlloc(d_ws, off, (size_t)EE * 4);
    int* maskA = (int*)wsAlloc(d_ws, off, (size_t)EE * 4);
    int* srcB  = (int*)wsAlloc(d_ws, off, (size_t)EE * 4);
    int* dstB  = (int*)wsAlloc(d_ws, off, (size_t)EE * 4);
    int* maskB = (int*)wsAlloc(d_ws, off, (size_t)EE * 4);
    float* attnm = (float*)wsAlloc(d_ws, off, (size_t)EE * 4);
    float* stats = (float*)wsAlloc(d_ws, off, 1024);
    float* pn    = (float*)wsAlloc(d_ws, off, 256);
    float* grep  = (float*)wsAlloc(d_ws, off, (size_t)BG * 2 * CC * 4);
    _Float16* wf16 = (_Float16*)wsAlloc(d_ws, off, 262144);   // swizzled weights

    auto gemm = [&](const float* A, const float* W_, const float* bias, float* O,
                    int M, int K, int Nc, int relu) {
        int npan = (K + 15) >> 4;
        int nbn  = Nc >> 6;
        if (K & 15) ZERO(wf16, (long)nbn * npan * 512);   // zero ragged K tail
        wswizzle_kernel<<<(K * Nc + 255) / 256, 256, 0, stream>>>(W_, wf16, K, Nc);
        int blocks = (M >> 7) * nbn;
        wmma_gemm_kernel<<<blocks, 256, 0, stream>>>(A, wf16, bias, O, M, K, Nc, relu);
    };

    ZERO(grep, BG * 2 * CC);

    const int* src = eidx;
    const int* dst = eidx + EE;
    const int* mask = nullptr;

    auto tconv = [&](const float* h, int n, const float* Wq_, const float* bq_,
                     const float* Wk_, const float* bk_, const float* Wv_, const float* bv_,
                     const float* We_, const float* Ws_, const float* bs_, const float* Wb_,
                     int Kin, bool meanOut) {
        gemm(h, Wq_, bq_, qb, n, Kin, HCHC, 0);
        gemm(h, Wk_, bk_, kb, n, Kin, HCHC, 0);
        gemm(h, Wv_, bv_, vb, n, Kin, HCHC, 0);
        gemm(h, Ws_, bs_, sb, n, Kin, HCHC, 0);
        ZERO(amax, (long)n * HH);
        attn_score_kernel<<<(EE * HH + 255) / 256, 256, 0, stream>>>(
            qb, kb, ea, We_, src, dst, mask, wE, amax, EE);
        ZERO(den, (long)n * HH);
        attn_exp_kernel<<<(EE * HH + 255) / 256, 256, 0, stream>>>(wE, amax, dst, mask, den, EE);
        attn_norm_kernel<<<(EE * HH + 255) / 256, 256, 0, stream>>>(wE, den, dst, EE);
        if (meanOut)
            attn_mean_kernel<<<(EE + 255) / 256, 256, 0, stream>>>(wE, attnm, EE);
        ZERO(aggb, (long)n * HCHC);
        aggregate_kernel<<<(EE * CC + 255) / 256, 256, 0, stream>>>(
            vb, ea, We_, wE, src, dst, mask, aggb, EE);
        betamix_kernel<<<(n + 255) / 256, 256, 0, stream>>>(aggb, sb, Wb_, n);
    };

    // ---- conv1: F_IN -> HC, then Wt0 -> 64, relu, BN ----
    tconv(x, N0, Wq0, bq0, Wk0, bk0, Wv0, bv0, We0, Ws0, bs0, Wb0, FIN, false);
    gemm(aggb, Wt0, bt0, h64a, N0, HCHC, CC, 1);
    ZERO(stats, 2 * CC);
    bn_stats_kernel<<<256, 256, 0, stream>>>(h64a, stats, N0);
    bn_apply_kernel<<<(N0 * CC + 255) / 256, 256, 0, stream>>>(h64a, stats, g0, b0, N0);

    // ---- stacked layers with pooling ----
    float* hcur = h64a;
    float* hnext = h64b;
    int n = N0, npg = NPG0;
    for (int i = 0; i < LNUM; ++i) {
        tconv(hcur, n, Wq + (size_t)i * CC * HCHC, bq + (size_t)i * HCHC,
              Wk + (size_t)i * CC * HCHC, bk + (size_t)i * HCHC,
              Wv + (size_t)i * CC * HCHC, bv + (size_t)i * HCHC,
              We + (size_t)i * EDIM * HCHC,
              Ws + (size_t)i * CC * HCHC, bs + (size_t)i * HCHC,
              Wb + (size_t)i * 3 * HCHC, CC, i == 0);
        gemm(aggb, Wt + (size_t)i * HCHC * CC, bt + (size_t)i * CC, hcur, n, HCHC, CC, 1);
        ZERO(stats, 2 * CC);
        bn_stats_kernel<<<256, 256, 0, stream>>>(hcur, stats, n);
        bn_apply_kernel<<<(n * CC + 255) / 256, 256, 0, stream>>>(
            hcur, stats, gg + (size_t)i * CC, bb + (size_t)i * CC, n);

        // TopK pooling (ratio 0.5)
        int k = npg / 2;
        pnorm_kernel<<<1, CC, 0, stream>>>(P + (size_t)i * CC, pn);
        score_kernel<<<(n + 255) / 256, 256, 0, stream>>>(hcur, P + (size_t)i * CC, pn,
                                                          score, remap, n);
        topk_kernel<<<BG, NPG0, 0, stream>>>(hcur, score, hnext, remap, npg, k);
        int* nsrc  = (i & 1) ? srcB : srcA;
        int* ndst  = (i & 1) ? dstB : dstA;
        int* nmask = (i & 1) ? maskB : maskA;
        remap_edges_kernel<<<(EE + 255) / 256, 256, 0, stream>>>(src, dst, mask, remap,
                                                                 nsrc, ndst, nmask, EE);
        src = nsrc; dst = ndst; mask = nmask;
        readout_kernel<<<(BG * CC + 255) / 256, 256, 0, stream>>>(hnext, grep, k);
        float* tmp = hcur; hcur = hnext; hnext = tmp;
        npg = k; n = BG * npg;
    }

    int total = BG * 2 * CC + 2 * EE + EE;
    finalize_kernel<<<(total + 255) / 256, 256, 0, stream>>>(grep, eidx, attnm, (float*)d_out);
}
